// DeltaBlock_78245714199337
// MI455X (gfx1250) — matrix operand order repaired and checked by reading, not compiled, not run on hardware
//
#include <hip/hip_runtime.h>
#include <hip/hip_bf16.h>

#define DIM 512
#define HID 1024
#define NPIX 65536
#define PPB  4096   // pixels per batch (64*64)

typedef __attribute__((ext_vector_type(16))) __bf16 v16bf;
typedef __attribute__((ext_vector_type(8)))  __bf16 v8bf;
typedef __attribute__((ext_vector_type(8)))  float  v8f;

// ---------- small device helpers ----------
__device__ __forceinline__ float wred32(float v) {
#pragma unroll
  for (int off = 16; off >= 1; off >>= 1) v += __shfl_xor(v, off, 32);
  return v;
}

__device__ __forceinline__ unsigned short f2bf(float f) {
  union { float f; unsigned u; } v; v.f = f;
  unsigned r = v.u + 0x7FFFu + ((v.u >> 16) & 1u);   // round-to-nearest-even
  return (unsigned short)(r >> 16);
}

// A fragment (16x32 bf16, row-major source, stride in elements), base pre-offset by k0.
// lane 0-15 : row=lane,    K = k0..k0+15
// lane 16-31: row=lane-16, K = k0+16..k0+31
__device__ __forceinline__ v16bf load_a_frag(const unsigned short* base, int stride, int lane) {
  const int r = lane & 15, half = lane >> 4;
  const unsigned short* p = base + r * stride + half * 16;
  v8bf lo = *(const v8bf*)(p);
  v8bf hi = *(const v8bf*)(p + 8);
  v16bf out;
#pragma unroll
  for (int i = 0; i < 8; ++i) { out[i] = lo[i]; out[i + 8] = hi[i]; }
  return out;
}

// B fragment (32x16 bf16) from K-major ("transposed") weights Bt[N][K], stride in elements.
// lane 0-15 : col = col0+lane, K = k0..k0+15 ; lanes 16-31: same col, K = k0+16..k0+31
__device__ __forceinline__ v16bf load_b_frag(const unsigned short* bt, int stride,
                                             int col0, int k0, int lane) {
  const int n = lane & 15, half = lane >> 4;
  const unsigned short* p = bt + (size_t)(col0 + n) * stride + k0 + half * 16;
  return *(const v16bf*)p;
}

// ---------- kernel 1: LayerNorm1 -> h1 ----------
__global__ void __launch_bounds__(256) k_ln1(const float* __restrict__ x,
                                             const float* __restrict__ w1,
                                             const float* __restrict__ b1,
                                             float* __restrict__ h1) {
  const int lane = threadIdx.x & 31, wave = threadIdx.x >> 5;
  const size_t pix = (size_t)blockIdx.x * 8 + wave;
  const float* xr = x + pix * DIM;
  float v[16]; float s1 = 0.f;
#pragma unroll
  for (int j = 0; j < 16; ++j) { v[j] = xr[lane + 32 * j]; s1 += v[j]; }
  s1 = wred32(s1);
  const float m = s1 * (1.0f / 512.0f);
  float s2 = 0.f;
#pragma unroll
  for (int j = 0; j < 16; ++j) { float d = v[j] - m; s2 += d * d; }
  s2 = wred32(s2);
  const float rs = rsqrtf(s2 * (1.0f / 512.0f) + 1e-5f);
  float* hr = h1 + pix * DIM;
#pragma unroll
  for (int j = 0; j < 16; ++j) {
    const int c = lane + 32 * j;
    hr[c] = (v[j] - m) * rs * w1[c] + b1[c];
  }
}

// ---------- kernel 2: per (batch,channel) sum of h1 over H*W ----------
__global__ void __launch_bounds__(256) k_gp(const float* __restrict__ h1,
                                            float* __restrict__ gp) {
  const int b = blockIdx.x;
  const int c = blockIdx.y * 256 + threadIdx.x;
  const float* p = h1 + (size_t)b * PPB * DIM + c;
  float s = 0.f;
  for (int t = 0; t < PPB; ++t) s += p[(size_t)t * DIM];
  gp[b * DIM + c] = s;
}

// ---------- kernel 3: normalize + transpose weights to bf16 ----------
__global__ void __launch_bounds__(256) k_wnorm(const float* __restrict__ pin,
                                               const float* __restrict__ pout,
                                               unsigned short* __restrict__ w_inT,
                                               unsigned short* __restrict__ w_outT) {
  const int lane = threadIdx.x & 31, wave = threadIdx.x >> 5;
  const int col = blockIdx.x * 8 + wave;
  if (col < HID) {                       // proto_in column: 512 elems, row stride HID
    float v[16]; float s = 0.f;
#pragma unroll
    for (int j = 0; j < 16; ++j) {
      const int k = lane + 32 * j;
      v[j] = pin[(size_t)k * HID + col]; s += v[j] * v[j];
    }
    s = wred32(s);
    const float inv = 1.0f / fmaxf(sqrtf(s), 1e-12f);
#pragma unroll
    for (int j = 0; j < 16; ++j)
      w_inT[(size_t)col * DIM + lane + 32 * j] = f2bf(v[j] * inv);
  } else {                               // proto_out column: 1024 elems, row stride DIM
    const int c2 = col - HID;
    float v[32]; float s = 0.f;
#pragma unroll
    for (int j = 0; j < 32; ++j) {
      const int k = lane + 32 * j;
      v[j] = pout[(size_t)k * DIM + c2]; s += v[j] * v[j];
    }
    s = wred32(s);
    const float inv = 1.0f / fmaxf(sqrtf(s), 1e-12f);
#pragma unroll
    for (int j = 0; j < 32; ++j)
      w_outT[(size_t)c2 * HID + lane + 32 * j] = f2bf(v[j] * inv);
  }
}

// ---------- kernel 4: spatial mix + residual -> d_out (= x2 shortcut) ----------
__global__ void __launch_bounds__(256) k_spatial(const float* __restrict__ x,
                                                 const float* __restrict__ h1,
                                                 const float* __restrict__ gp,
                                                 const float* __restrict__ alpha,
                                                 const float* __restrict__ gamma,
                                                 float* __restrict__ xout) {
  const int lane = threadIdx.x & 31, wave = threadIdx.x >> 5;
  const int pix = blockIdx.x * 8 + wave;
  const int b = pix >> 12, rem = pix & 4095, hh = rem >> 6, ww = rem & 63;
  const float* c0 = h1 + (size_t)pix * DIM;
  const float* up = (hh > 0)  ? c0 - 64 * DIM : nullptr;
  const float* dn = (hh < 63) ? c0 + 64 * DIM : nullptr;
  const float* lf = (ww > 0)  ? c0 - DIM      : nullptr;
  const float* rt = (ww < 63) ? c0 + DIM      : nullptr;
  const float* xr = x + (size_t)pix * DIM;
  float* orow = xout + (size_t)pix * DIM;
#pragma unroll
  for (int j = 0; j < 16; ++j) {
    const int c = lane + 32 * j;
    const float ce = c0[c];
    float fe = fabsf(ce - (up ? up[c] : 0.f)) + fabsf(ce - (dn ? dn[c] : 0.f)) +
               fabsf(ce - (lf ? lf[c] : 0.f)) + fabsf(ce - (rt ? rt[c] : 0.f));
    const float gpm = gp[b * DIM + c] * (1.0f / 4096.0f);
    const float al = alpha[c];
    const float sp = fe * al + gpm * (1.0f - al);
    orow[c] = xr[c] + gamma[c] * sp;
  }
}

// ---------- kernel 5: fused LN2 + L2 + GEMM1 + GELU + L2 + GEMM2 + residual ----------
// 32 rows per block: each wave owns TWO 16-row M-tiles, so every B fragment
// loaded from L2 feeds two WMMAs (halves dominant weight traffic to ~4 GB).
__global__ void __launch_bounds__(256) k_channel(float* __restrict__ xio,
                                                 const float* __restrict__ w2,
                                                 const float* __restrict__ b2,
                                                 const unsigned short* __restrict__ w_inT,
                                                 const unsigned short* __restrict__ w_outT,
                                                 const float* __restrict__ scale_in,
                                                 const float* __restrict__ scale_out,
                                                 const float* __restrict__ gamma) {
  __shared__ alignas(16) unsigned short xn_lds[32 * DIM];   // 32 KB bf16 activations
  __shared__ alignas(16) unsigned short hid_lds[32 * HID];  // 64 KB bf16 hidden
  __shared__ float rowsq[32];

  const int tid = threadIdx.x;
  const int lane = tid & 31;
  const int wave = tid >> 5;
  const int row0 = blockIdx.x * 32;
  const int half = lane >> 4;
  const int nidx = lane & 15;

  if (tid < 32) rowsq[tid] = 0.0f;
  const float s_in = scale_in[0];
  const float s_out = scale_out[0];

  // ---- stage 1: LayerNorm2 + row L2 norm -> bf16 in LDS (4 rows per wave) ----
#pragma unroll
  for (int rr = 0; rr < 4; ++rr) {
    const int r = wave * 4 + rr;
    const float* xr = xio + (size_t)(row0 + r) * DIM;
    float v[16]; float s1 = 0.f;
#pragma unroll
    for (int j = 0; j < 16; ++j) { v[j] = xr[lane + 32 * j]; s1 += v[j]; }
    s1 = wred32(s1);
    const float m = s1 * (1.0f / 512.0f);
    float s2 = 0.f;
#pragma unroll
    for (int j = 0; j < 16; ++j) { float d = v[j] - m; s2 += d * d; }
    s2 = wred32(s2);
    const float rs = rsqrtf(s2 * (1.0f / 512.0f) + 1e-5f);
    float ss = 0.f;
#pragma unroll
    for (int j = 0; j < 16; ++j) {
      const int c = lane + 32 * j;
      const float h = (v[j] - m) * rs * w2[c] + b2[c];
      v[j] = h; ss += h * h;
    }
    ss = wred32(ss);
    const float inv = 1.0f / fmaxf(sqrtf(ss), 1e-12f);
#pragma unroll
    for (int j = 0; j < 16; ++j)
      xn_lds[r * DIM + lane + 32 * j] = f2bf(v[j] * inv);
  }
  __syncthreads();

  // ---- GEMM1: hidden(32x1024) = xn(32x512) @ w_in ; wave: 2 row-tiles x 8 col-tiles ----
  v8f acc1[2][8] = {};
  const int ct0 = wave * 8;
  for (int kk = 0; kk < DIM; kk += 32) {
    const v16bf a0 = load_a_frag(xn_lds + kk, DIM, lane);
    const v16bf a1 = load_a_frag(xn_lds + 16 * DIM + kk, DIM, lane);
#pragma unroll
    for (int i = 0; i < 8; ++i) {
      const v16bf b = load_b_frag(w_inT, DIM, (ct0 + i) * 16, kk, lane);
      acc1[0][i] = __builtin_amdgcn_wmma_f32_16x16x32_bf16(false, a0, false, b,
                                                           (short)0, acc1[0][i], false, false);
      acc1[1][i] = __builtin_amdgcn_wmma_f32_16x16x32_bf16(false, a1, false, b,
                                                           (short)0, acc1[1][i], false, false);
    }
  }

  // ---- exact GELU + per-row sum of squares ----
#pragma unroll
  for (int rt = 0; rt < 2; ++rt) {
    float part[8];
#pragma unroll
    for (int g = 0; g < 8; ++g) part[g] = 0.f;
#pragma unroll
    for (int i = 0; i < 8; ++i) {
#pragma unroll
      for (int g = 0; g < 8; ++g) {
        const float xv = acc1[rt][i][g] * s_in;
        const float gv = 0.5f * xv * (1.0f + erff(xv * 0.70710678f));
        acc1[rt][i][g] = gv;
        part[g] += gv * gv;
      }
    }
#pragma unroll
    for (int g = 0; g < 8; ++g) atomicAdd(&rowsq[rt * 16 + g + 8 * half], part[g]);
  }
  __syncthreads();

  // ---- row L2 normalize hidden -> bf16 in LDS ----
#pragma unroll
  for (int rt = 0; rt < 2; ++rt) {
#pragma unroll
    for (int g = 0; g < 8; ++g) {
      const int r = rt * 16 + g + 8 * half;
      const float inv = 1.0f / fmaxf(sqrtf(rowsq[r]), 1e-12f);
#pragma unroll
      for (int i = 0; i < 8; ++i) {
        const int col = (ct0 + i) * 16 + nidx;
        hid_lds[r * HID + col] = f2bf(acc1[rt][i][g] * inv);
      }
    }
  }
  __syncthreads();

  // ---- GEMM2: out(32x512) = hn(32x1024) @ w_out ; wave: 2 row-tiles x 4 col-tiles ----
  v8f acc2[2][4] = {};
  const int ct2 = wave * 4;
  for (int kk = 0; kk < HID; kk += 32) {
    const v16bf a0 = load_a_frag(hid_lds + kk, HID, lane);
    const v16bf a1 = load_a_frag(hid_lds + 16 * HID + kk, HID, lane);
#pragma unroll
    for (int i = 0; i < 4; ++i) {
      const v16bf b = load_b_frag(w_outT, HID, (ct2 + i) * 16, kk, lane);
      acc2[0][i] = __builtin_amdgcn_wmma_f32_16x16x32_bf16(false, a0, false, b,
                                                           (short)0, acc2[0][i], false, false);
      acc2[1][i] = __builtin_amdgcn_wmma_f32_16x16x32_bf16(false, a1, false, b,
                                                           (short)0, acc2[1][i], false, false);
    }
  }

  // ---- epilogue: scale_out, gamma, residual (d_out read-modify-write) ----
#pragma unroll
  for (int i = 0; i < 4; ++i) {
    const int col = (ct2 + i) * 16 + nidx;
    const float gm = gamma[col];
#pragma unroll
    for (int rt = 0; rt < 2; ++rt) {
#pragma unroll
      for (int g = 0; g < 8; ++g) {
        const int r = rt * 16 + g + 8 * half;
        const size_t idx = (size_t)(row0 + r) * DIM + col;
        xio[idx] = xio[idx] + gm * (acc2[rt][i][g] * s_out);
      }
    }
  }
}

// ---------- host launcher ----------
extern "C" void kernel_launch(void* const* d_in, const int* in_sizes, int n_in,
                              void* d_out, int out_size, void* d_ws, size_t ws_size,
                              hipStream_t stream) {
  (void)in_sizes; (void)n_in; (void)out_size; (void)ws_size;
  const float* x     = (const float*)d_in[0];
  const float* n1w   = (const float*)d_in[1];
  const float* n1b   = (const float*)d_in[2];
  const float* alpha = (const float*)d_in[3];
  const float* n2w   = (const float*)d_in[4];
  const float* n2b   = (const float*)d_in[5];
  const float* pin   = (const float*)d_in[6];
  const float* pout  = (const float*)d_in[7];
  const float* s_in  = (const float*)d_in[8];
  const float* s_out = (const float*)d_in[9];
  const float* gamma = (const float*)d_in[10];
  float* out = (float*)d_out;

  // workspace carve-up
  float* h1 = (float*)d_ws;                                 // 65536*512 f32 (128 MB)
  float* gp = h1 + (size_t)NPIX * DIM;                      // 16*512 f32
  unsigned short* w_inT  = (unsigned short*)(gp + 16 * DIM);// 1024*512 bf16 (1 MB)
  unsigned short* w_outT = w_inT + (size_t)HID * DIM;       // 512*1024 bf16 (1 MB)

  k_ln1    <<<NPIX / 8, 256, 0, stream>>>(x, n1w, n1b, h1);
  k_wnorm  <<<192, 256, 0, stream>>>(pin, pout, w_inT, w_outT);
  k_gp     <<<dim3(16, 2), 256, 0, stream>>>(h1, gp);
  k_spatial<<<NPIX / 8, 256, 0, stream>>>(x, h1, gp, alpha, gamma, out);
  k_channel<<<NPIX / 32, 256, 0, stream>>>(out, n2w, n2b, w_inT, w_outT, s_in, s_out, gamma);
}
